// GroupedQueryAttention_20607253086681
// MI455X (gfx1250) — matrix operand order repaired and checked by reading, not compiled, not run on hardware
//
#include <hip/hip_runtime.h>
#include <hip/hip_bf16.h>
#include <math.h>

// ---------------------------------------------------------------------------
// Types for CDNA5 WMMA (wave32): A = 16x32 bf16, B = 32x16 bf16, C/D = 16x16 f32
// ---------------------------------------------------------------------------
typedef __bf16 bf16;
typedef __attribute__((ext_vector_type(16))) __bf16 v16bf;
typedef __attribute__((ext_vector_type(8)))  __bf16 v8bf;
typedef __attribute__((ext_vector_type(4)))  __bf16 v4bf;
typedef __attribute__((ext_vector_type(8)))  float  v8f;
typedef __attribute__((ext_vector_type(4)))  float  v4f;
typedef __attribute__((ext_vector_type(4)))  int    v4i;

// Async global->LDS (CDNA5 GLOBAL_LOAD_ASYNC_TO_LDS_*, tracked by ASYNCcnt)
#if defined(__has_builtin)
#  if __has_builtin(__builtin_amdgcn_global_load_async_to_lds_b128) && \
      __has_builtin(__builtin_amdgcn_s_wait_asynccnt)
#    define HAS_ASYNC_LDS 1
#  endif
#endif
#ifndef HAS_ASYNC_LDS
#  define HAS_ASYNC_LDS 0
#endif

static __device__ inline v8f zero_v8f() {
    v8f z;
#pragma unroll
    for (int i = 0; i < 8; ++i) z[i] = 0.f;
    return z;
}

static __device__ inline v8f wmma_bf16(v16bf a, v16bf b, v8f c) {
    // D = A(16x32) * B(32x16) + C ; emits v_wmma_f32_16x16x32_bf16
    return __builtin_amdgcn_wmma_f32_16x16x32_bf16(
        /*neg_a=*/false, a, /*neg_b=*/false, b,
        /*c_mod=*/(short)0, c, /*reuse_a=*/false, /*reuse_b=*/false);
}

#if HAS_ASYNC_LDS
static __device__ __forceinline__ void async_cp16(void* lds, const void* g) {
    // signature (from clang diagnostic): param0 = int4 addrspace(1)*
    __builtin_amdgcn_global_load_async_to_lds_b128(
        (__attribute__((address_space(1))) v4i*)g,
        (__attribute__((address_space(3))) v4i*)lds,
        /*offset=*/0, /*cpol=*/0);
}
#endif

#define NEG_INF (-1e30f)

// ---------------------------------------------------------------------------
// fp32 -> bf16 conversion (vectorized x4)
// ---------------------------------------------------------------------------
__global__ void __launch_bounds__(256)
cvt_bf16_kernel(const float* __restrict__ in, bf16* __restrict__ out, int n4) {
    int i = blockIdx.x * 256 + threadIdx.x;
    if (i >= n4) return;
    v4f v = *(const v4f*)(in + (size_t)i * 4);
    v4bf o;
    o.x = (bf16)v.x; o.y = (bf16)v.y; o.z = (bf16)v.z; o.w = (bf16)v.w;
    *(v4bf*)(out + (size_t)i * 4) = o;
}

// ---------------------------------------------------------------------------
// GEMM: Y[M][N] = X[M][K] * W[N][K]^T (+bias).  bf16 inputs, fp32 output.
// Block tile 128x128, 8 waves (4 along M x 2 along N), wave tile 32x64,
// K-step 32.  LDS double buffer (+8 elem pad keeps rows 16B aligned).
// Async path: each wave issues 4 GLOBAL_LOAD_ASYNC_TO_LDS_B128 per K-tile,
// waits ASYNCcnt<=4 (previous group done, in-order), overlapping DMA w/ WMMA.
// ---------------------------------------------------------------------------
static __device__ __forceinline__ void
gemm_compute_tile(const bf16 (*bx)[40], const bf16 (*bw)[40],
                  int mw, int nw, int l16, int lh, v8f (&acc)[2][4]) {
    // A fragments: lane l16 = row M; halves carry K 0..15 | 16..31
    v16bf afrag[2];
#pragma unroll
    for (int ms = 0; ms < 2; ++ms) {
        int row = mw * 32 + ms * 16 + l16;
        ((v8bf*)&afrag[ms])[0] = *(const v8bf*)&bx[row][lh * 16];
        ((v8bf*)&afrag[ms])[1] = *(const v8bf*)&bx[row][lh * 16 + 8];
    }
    // B fragments: lane l16 = col N; halves carry K 0..15 | 16..31
    v16bf bfrag[4];
#pragma unroll
    for (int ns = 0; ns < 4; ++ns) {
        int row = nw * 64 + ns * 16 + l16;
        ((v8bf*)&bfrag[ns])[0] = *(const v8bf*)&bw[row][lh * 16];
        ((v8bf*)&bfrag[ns])[1] = *(const v8bf*)&bw[row][lh * 16 + 8];
    }
#pragma unroll
    for (int ms = 0; ms < 2; ++ms)
#pragma unroll
        for (int ns = 0; ns < 4; ++ns)
            acc[ms][ns] = wmma_bf16(afrag[ms], bfrag[ns], acc[ms][ns]);
}

__global__ void __launch_bounds__(256)
gemm_bf16_wmma(const bf16* __restrict__ X, const bf16* __restrict__ W,
               float* __restrict__ Y, const float* __restrict__ bias,
               int M, int N, int K) {
    __shared__ bf16 lX[2][128][40];
    __shared__ bf16 lW[2][128][40];

    const int tid  = threadIdx.x;
    const int lane = tid & 31;
    const int wid  = tid >> 5;       // 0..7
    const int mw   = wid & 3;        // 4 waves along M (32 rows each)
    const int nw   = wid >> 2;       // 2 waves along N (64 cols each)
    const int m0   = blockIdx.y * 128;
    const int n0   = blockIdx.x * 128;
    const int lh   = lane >> 4;      // 0/1 half of wave
    const int l16  = lane & 15;

    v8f acc[2][4];
#pragma unroll
    for (int i = 0; i < 2; ++i)
#pragma unroll
        for (int j = 0; j < 4; ++j) acc[i][j] = zero_v8f();

    // staging assignment: 256 threads, 16 bf16 (32B) per tile each
    const int lrow = tid >> 1;          // 0..127
    const int lcol = (tid & 1) * 16;    // 0 or 16

    const bf16* gx = X + (size_t)(m0 + lrow) * K + lcol;
    const bf16* gw = W + (size_t)(n0 + lrow) * K + lcol;

#if HAS_ASYNC_LDS
    // ---------------- async DMA double-buffered pipeline -------------------
    {
        // prologue: group 0 -> buffer 0   (4 async instr per wave)
        async_cp16(&lX[0][lrow][lcol],     gx);
        async_cp16(&lX[0][lrow][lcol + 8], gx + 8);
        async_cp16(&lW[0][lrow][lcol],     gw);
        async_cp16(&lW[0][lrow][lcol + 8], gw + 8);
    }
    for (int k0 = 0, it = 0; k0 < K; k0 += 32, ++it) {
        const int cur = it & 1, nxt = cur ^ 1;
        const bool more = (k0 + 32 < K);
        if (more) {   // issue group it+1 into the other buffer
            async_cp16(&lX[nxt][lrow][lcol],     gx + k0 + 32);
            async_cp16(&lX[nxt][lrow][lcol + 8], gx + k0 + 40);
            async_cp16(&lW[nxt][lrow][lcol],     gw + k0 + 32);
            async_cp16(&lW[nxt][lrow][lcol + 8], gw + k0 + 40);
            __builtin_amdgcn_s_wait_asynccnt(4);   // group `it` landed
        } else {
            __builtin_amdgcn_s_wait_asynccnt(0);
        }
        __syncthreads();                            // all waves' group landed
        gemm_compute_tile(lX[cur], lW[cur], mw, nw, l16, lh, acc);
        __syncthreads();                            // done reading `cur`
    }
#else
    // ---------------- register-prefetch software pipeline ------------------
    uint4 x0, x1, w0, w1;
    x0 = ((const uint4*)gx)[0]; x1 = ((const uint4*)gx)[1];
    w0 = ((const uint4*)gw)[0]; w1 = ((const uint4*)gw)[1];
    for (int k0 = 0, it = 0; k0 < K; k0 += 32, ++it) {
        const int cur = it & 1;
        __syncthreads();                 // readers of `cur` (it-2) are done
        *(uint4*)&lX[cur][lrow][lcol]     = x0;
        *(uint4*)&lX[cur][lrow][lcol + 8] = x1;
        *(uint4*)&lW[cur][lrow][lcol]     = w0;
        *(uint4*)&lW[cur][lrow][lcol + 8] = w1;
        if (k0 + 32 < K) {               // prefetch tile it+1 into registers
            x0 = ((const uint4*)(gx + k0 + 32))[0];
            x1 = ((const uint4*)(gx + k0 + 32))[1];
            w0 = ((const uint4*)(gw + k0 + 32))[0];
            w1 = ((const uint4*)(gw + k0 + 32))[1];
            if (k0 + 64 < K) {
                __builtin_prefetch(gx + k0 + 64, 0, 1);
                __builtin_prefetch(gw + k0 + 64, 0, 1);
            }
        }
        __syncthreads();
        gemm_compute_tile(lX[cur], lW[cur], mw, nw, l16, lh, acc);
    }
#endif

    // C/D layout: lane l16 = col, reg r -> row r + 8*lh
#pragma unroll
    for (int ms = 0; ms < 2; ++ms)
#pragma unroll
        for (int ns = 0; ns < 4; ++ns) {
            int col = n0 + nw * 64 + ns * 16 + l16;
            float bv = bias ? bias[col] : 0.f;
#pragma unroll
            for (int r = 0; r < 8; ++r) {
                int row = m0 + mw * 32 + ms * 16 + r + 8 * lh;
                Y[(size_t)row * N + col] = acc[ms][ns][r] + bv;
            }
        }
}

// ---------------------------------------------------------------------------
// Fused RMSNorm + RoPE + repack:  Yf [B,T,nh,128] f32  ->  out [B,nh,T,128] bf16
// One wave per (b,t,head) row; lane owns d = lane, lane+32, lane+64, lane+96.
// RoPE pairs (d, d+64) live in the same lane (j=0<->2, j=1<->3).
// oscale folds 1/sqrt(head_dim) for Q.
// ---------------------------------------------------------------------------
__global__ void __launch_bounds__(256)
norm_rope_pack(const float* __restrict__ Yf, const float* __restrict__ g,
               const float* __restrict__ cs, const float* __restrict__ sn,
               bf16* __restrict__ out, int nh, int T, float oscale) {
    const int lane = threadIdx.x & 31;
    const int wid  = threadIdx.x >> 5;
    const int row  = blockIdx.x * 8 + wid;      // in [0, B*T*nh)
    const int hh = row % nh;
    const int bt = row / nh;                    // b*T + t
    const int t  = bt % T;
    const int b  = bt / T;

    const float* src = Yf + (size_t)row * 128;
    float x[4];
    float ss = 0.f;
#pragma unroll
    for (int j = 0; j < 4; ++j) { x[j] = src[lane + 32 * j]; ss += x[j] * x[j]; }
    ss += __shfl_xor(ss, 1);  ss += __shfl_xor(ss, 2);  ss += __shfl_xor(ss, 4);
    ss += __shfl_xor(ss, 8);  ss += __shfl_xor(ss, 16);
    float rinv = rsqrtf(ss * (1.f / 128.f) + 1e-6f);

    float xn[4];
#pragma unroll
    for (int j = 0; j < 4; ++j) {
        int d = lane + 32 * j;
        xn[j] = x[j] * rinv * g[d];
    }
    bf16* dst = out + ((size_t)(b * nh + hh) * T + t) * 128;
#pragma unroll
    for (int j = 0; j < 4; ++j) {
        int d = lane + 32 * j;
        float rot = (j < 2) ? -xn[j + 2] : xn[j - 2];
        float v = xn[j] * cs[(size_t)t * 128 + d] + rot * sn[(size_t)t * 128 + d];
        dst[d] = (bf16)(v * oscale);
    }
}

// ---------------------------------------------------------------------------
// V pack + transpose:  Vf [B,T,8,128] f32  ->  Vt [B,8,128,T] bf16
// ---------------------------------------------------------------------------
__global__ void __launch_bounds__(256)
v_pack(const float* __restrict__ Vf, bf16* __restrict__ Vt, int T) {
    __shared__ float tile[32][33];
    const int bkv = blockIdx.z;           // b*8 + kv
    const int d0  = blockIdx.y * 32;
    const int t0  = blockIdx.x * 32;
    const int b   = bkv >> 3, kv = bkv & 7;
    const int tx  = threadIdx.x & 31, ty = threadIdx.x >> 5;   // ty 0..7
    for (int i = ty; i < 32; i += 8)
        tile[i][tx] = Vf[(((size_t)(b * T + t0 + i)) * 8 + kv) * 128 + d0 + tx];
    __syncthreads();
    for (int i = ty; i < 32; i += 8)
        Vt[((size_t)bkv * 128 + d0 + i) * T + t0 + tx] = (bf16)tile[tx][i];
}

// ---------------------------------------------------------------------------
// Flash attention, causal, GQA (rep=2).  One wave per 16 query rows.
// Q [B,16,T,128], K [B,8,T,128], Vt [B,8,128,T], all bf16 (Q pre-scaled).
// Output Obf [B,T,2048] bf16 (== 'bqhd' reshape, ready for O-projection GEMM).
// ---------------------------------------------------------------------------
__global__ void __launch_bounds__(128)
flash_attn(const bf16* __restrict__ Q, const bf16* __restrict__ Kb,
           const bf16* __restrict__ Vt, bf16* __restrict__ O, int T) {
    __shared__ bf16 lP[4][16][40];        // per-wave P transpose staging
    const int lane = threadIdx.x & 31;
    const int wid  = threadIdx.x >> 5;
    const int l16  = lane & 15, lh = lane >> 4;
    const int q0   = blockIdx.x * 64 + wid * 16;
    const int h    = blockIdx.y, b = blockIdx.z;
    const int kv   = h >> 1;              // rep = N_HEADS / N_KV = 2

    const bf16* Qp = Q  + ((size_t)(b * 16 + h) * T + q0) * 128;
    const bf16* Kp = Kb + (size_t)(b * 8 + kv) * T * 128;
    const bf16* Vp = Vt + (size_t)(b * 8 + kv) * 128 * T;

    // Persistent Q A-fragments: 4 K-chunks of 32 along head_dim
    v16bf qf[4];
#pragma unroll
    for (int dblk = 0; dblk < 4; ++dblk) {
        const bf16* base = Qp + (size_t)l16 * 128 + dblk * 32;
        ((v8bf*)&qf[dblk])[0] = *(const v8bf*)(base + lh * 16);
        ((v8bf*)&qf[dblk])[1] = *(const v8bf*)(base + lh * 16 + 8);
    }

    v8f o_acc[8];
#pragma unroll
    for (int dt = 0; dt < 8; ++dt) o_acc[dt] = zero_v8f();
    float m_i[8], l_i[8];
#pragma unroll
    for (int r = 0; r < 8; ++r) { m_i[r] = NEG_INF; l_i[r] = 0.f; }

    for (int t0 = 0; t0 < q0 + 16; t0 += 32) {
        // hint next K/V block toward the caches while we crunch this one
        if (t0 + 32 < q0 + 16) {
            __builtin_prefetch(Kp + (size_t)(t0 + 32 + l16) * 128, 0, 1);
            __builtin_prefetch(Vp + (size_t)l16 * T + t0 + 32, 0, 1);
        }
        // S = Q * K^T for 32 keys (two 16x16 tiles), accumulate over d
        v8f s0 = zero_v8f(), s1 = zero_v8f();
#pragma unroll
        for (int dblk = 0; dblk < 4; ++dblk) {
            v16bf b0, b1;
            const bf16* kb0 = Kp + (size_t)(t0 + l16) * 128 + dblk * 32 + lh * 16;
            const bf16* kb1 = Kp + (size_t)(t0 + 16 + l16) * 128 + dblk * 32 + lh * 16;
            ((v8bf*)&b0)[0] = *(const v8bf*)kb0;
            ((v8bf*)&b0)[1] = *(const v8bf*)(kb0 + 8);
            ((v8bf*)&b1)[0] = *(const v8bf*)kb1;
            ((v8bf*)&b1)[1] = *(const v8bf*)(kb1 + 8);
            s0 = wmma_bf16(qf[dblk], b0, s0);
            s1 = wmma_bf16(qf[dblk], b1, s1);
        }
        // causal mask: only diagonal blocks need it
        if (t0 + 31 > q0) {
#pragma unroll
            for (int r = 0; r < 8; ++r) {
                int qg = q0 + r + 8 * lh;
                if (t0 + l16 > qg)      s0[r] = NEG_INF;
                if (t0 + 16 + l16 > qg) s1[r] = NEG_INF;
            }
        }
        // online softmax (fp32 stats; row reductions within 16-lane halves)
        float pscale[8], p0[8], p1[8];
#pragma unroll
        for (int r = 0; r < 8; ++r) {
            float mx = fmaxf(s0[r], s1[r]);
            mx = fmaxf(mx, __shfl_xor(mx, 1));
            mx = fmaxf(mx, __shfl_xor(mx, 2));
            mx = fmaxf(mx, __shfl_xor(mx, 4));
            mx = fmaxf(mx, __shfl_xor(mx, 8));
            float mnew = fmaxf(m_i[r], mx);
            float sc   = __expf(m_i[r] - mnew);
            p0[r] = __expf(s0[r] - mnew);
            p1[r] = __expf(s1[r] - mnew);
            float rs = p0[r] + p1[r];
            rs += __shfl_xor(rs, 1); rs += __shfl_xor(rs, 2);
            rs += __shfl_xor(rs, 4); rs += __shfl_xor(rs, 8);
            l_i[r]    = l_i[r] * sc + rs;
            m_i[r]    = mnew;
            pscale[r] = sc;
        }
#pragma unroll
        for (int dt = 0; dt < 8; ++dt)
#pragma unroll
            for (int r = 0; r < 8; ++r) o_acc[dt][r] *= pscale[r];

        // Transpose P (C/D layout) -> A layout via per-wave LDS tile
#pragma unroll
        for (int r = 0; r < 8; ++r) {
            lP[wid][r + 8 * lh][l16]      = (bf16)p0[r];
            lP[wid][r + 8 * lh][16 + l16] = (bf16)p1[r];
        }
        // same-wave LDS RAW: DS ops are in-order per wave; compiler waits DScnt
        v16bf pa;
        ((v8bf*)&pa)[0] = *(const v8bf*)&lP[wid][l16][lh * 16];
        ((v8bf*)&pa)[1] = *(const v8bf*)&lP[wid][l16][lh * 16 + 8];

        // O += P(16x32) * V(32x128) across 8 d-tiles
#pragma unroll
        for (int dt = 0; dt < 8; ++dt) {
            v16bf vb;
            const bf16* vbase = Vp + (size_t)(dt * 16 + l16) * T + t0 + lh * 16;
            ((v8bf*)&vb)[0] = *(const v8bf*)vbase;
            ((v8bf*)&vb)[1] = *(const v8bf*)(vbase + 8);
            o_acc[dt] = wmma_bf16(pa, vb, o_acc[dt]);
        }
    }

    // normalize + store: Obf [B,T,H*128] bf16 (bqhd layout)
    bf16* Op = O + ((size_t)b * T + q0) * 2048 + h * 128;
#pragma unroll
    for (int r = 0; r < 8; ++r) {
        float inv = 1.f / l_i[r];
#pragma unroll
        for (int dt = 0; dt < 8; ++dt)
            Op[(size_t)(r + 8 * lh) * 2048 + dt * 16 + l16] =
                (bf16)(o_acc[dt][r] * inv);
    }
}

// ---------------------------------------------------------------------------
// Host-side launch
// ---------------------------------------------------------------------------
extern "C" void kernel_launch(void* const* d_in, const int* in_sizes, int n_in,
                              void* d_out, int out_size, void* d_ws, size_t ws_size,
                              hipStream_t stream) {
    const float* x  = (const float*)d_in[0];
    // d_in[1] = mask (unused; causal mask computed analytically)
    const float* cs = (const float*)d_in[2];
    const float* sn = (const float*)d_in[3];
    const float* Wq = (const float*)d_in[4];
    const float* Wk = (const float*)d_in[5];
    const float* Wv = (const float*)d_in[6];
    const float* Wo = (const float*)d_in[7];
    const float* bo = (const float*)d_in[8];
    const float* gq = (const float*)d_in[9];
    const float* gk = (const float*)d_in[10];
    float* out = (float*)d_out;

    const int B = 2, T = 2048, D = 2048, H = 16, KV = 8;
    const int M = B * T;                    // 4096
    const float qscale = 0.08838834764831845f;   // 1/sqrt(128)

    // workspace layout (bytes)
    char* w = (char*)d_ws;
    bf16*  xbf = (bf16*)w;                 w += (size_t)M * D * 2;          // 16 MB
    bf16*  wqb = (bf16*)w;                 w += (size_t)D * D * 2;          //  8 MB
    bf16*  wkb = (bf16*)w;                 w += (size_t)(KV * 128) * D * 2; //  4 MB
    bf16*  wvb = (bf16*)w;                 w += (size_t)(KV * 128) * D * 2; //  4 MB
    bf16*  wob = (bf16*)w;                 w += (size_t)D * D * 2;          //  8 MB
    bf16*  qbf = (bf16*)w;                 w += (size_t)M * D * 2;          // 16 MB
    bf16*  kbf = (bf16*)w;                 w += (size_t)M * (KV * 128) * 2; //  8 MB
    bf16*  vtb = (bf16*)w;                 w += (size_t)M * (KV * 128) * 2; //  8 MB
    bf16*  obf = (bf16*)w;                 w += (size_t)M * D * 2;          // 16 MB
    float* f32s = (float*)w;               /* 32 MB fp32 GEMM scratch */

    // 1) convert activations + weights to bf16
    cvt_bf16_kernel<<<(M * D / 4 + 255) / 256, 256, 0, stream>>>(x,  xbf, M * D / 4);
    cvt_bf16_kernel<<<(D * D / 4 + 255) / 256, 256, 0, stream>>>(Wq, wqb, D * D / 4);
    cvt_bf16_kernel<<<(KV * 128 * D / 4 + 255) / 256, 256, 0, stream>>>(Wk, wkb, KV * 128 * D / 4);
    cvt_bf16_kernel<<<(KV * 128 * D / 4 + 255) / 256, 256, 0, stream>>>(Wv, wvb, KV * 128 * D / 4);
    cvt_bf16_kernel<<<(D * D / 4 + 255) / 256, 256, 0, stream>>>(Wo, wob, D * D / 4);

    // 2) Q projection + RMSNorm + RoPE (+1/sqrt(d) folded in)
    gemm_bf16_wmma<<<dim3(D / 128, M / 128), 256, 0, stream>>>(
        xbf, wqb, f32s, nullptr, M, D, D);
    norm_rope_pack<<<M * H / 8, 256, 0, stream>>>(f32s, gq, cs, sn, qbf, H, T, qscale);

    // 3) K projection + RMSNorm + RoPE
    gemm_bf16_wmma<<<dim3(KV * 128 / 128, M / 128), 256, 0, stream>>>(
        xbf, wkb, f32s, nullptr, M, KV * 128, D);
    norm_rope_pack<<<M * KV / 8, 256, 0, stream>>>(f32s, gk, cs, sn, kbf, KV, T, 1.0f);

    // 4) V projection + transpose to [B,KV,128,T]
    gemm_bf16_wmma<<<dim3(KV * 128 / 128, M / 128), 256, 0, stream>>>(
        xbf, wvb, f32s, nullptr, M, KV * 128, D);
    v_pack<<<dim3(T / 32, 128 / 32, B * KV), 256, 0, stream>>>(f32s, vtb, T);

    // 5) causal flash attention (GQA rep=2)
    flash_attn<<<dim3(T / 64, H, B), 128, 0, stream>>>(qbf, kbf, vtb, obf, T);

    // 6) output projection + bias -> fp32 d_out
    gemm_bf16_wmma<<<dim3(D / 128, M / 128), 256, 0, stream>>>(
        obf, wob, out, bo, M, D, D);
}
